// EdgeTypePredictionHetero_10462540333788
// MI455X (gfx1250) — matrix operand order repaired and checked by reading, not compiled, not run on hardware
//
#include <hip/hip_runtime.h>

// ---------------------------------------------------------------------------
// EdgeTypePredictionHetero fused kernel for gfx1250 (MI455X, CDNA5, wave32)
//   h       = relu([h_src|h_dst] @ Wcat + b_dec)       GEMM1: K=512, N=256
//   logitsA = h @ Wp_all (+ b_pred)                    GEMM2: K=256, N=256
//   select 16 cols for edge's (src,dst) pair, log-softmax CE, group-mean loss
// Matrix work on v_wmma_f32_16x16x32_bf16; weight chunks staged to LDS with
// global_load_async_to_lds_b128 (ASYNCcnt).
// ---------------------------------------------------------------------------

#define USE_ASYNC_LDS 1

typedef __attribute__((ext_vector_type(16))) __bf16 v16bf;
typedef __attribute__((ext_vector_type(8)))  float  v8f;

union Frag {
    v16bf v;
    uint4 q[2];
    unsigned int u[8];
};

__device__ __forceinline__ unsigned int rnd_bf(unsigned int u) {
    return u + 0x7FFFu + ((u >> 16) & 1u);            // RNE bias in high half
}
// pack bf16(a) | bf16(b)<<16 with one v_perm_b32
__device__ __forceinline__ unsigned int pk2bf(float a, float b) {
    unsigned int ua = rnd_bf(__float_as_uint(a));
    unsigned int ub = rnd_bf(__float_as_uint(b));
    return __builtin_amdgcn_perm(ub, ua, 0x07060302u);
}
__device__ __forceinline__ unsigned short f2bf(float f) {
    return (unsigned short)(rnd_bf(__float_as_uint(f)) >> 16);
}

__device__ __forceinline__ void wait_async() {
#if __has_builtin(__builtin_amdgcn_s_wait_asynccnt)
    __builtin_amdgcn_s_wait_asynccnt(0);
#else
    asm volatile("s_wait_asynccnt 0x0" ::: "memory");
#endif
}

// copy 16B global -> LDS, tracked by ASYNCcnt (no VGPR round-trip)
__device__ __forceinline__ void async_copy16(unsigned lds_off, const void* gptr) {
#if USE_ASYNC_LDS
    unsigned long long ga = (unsigned long long)(uintptr_t)gptr;
    asm volatile("global_load_async_to_lds_b128 %0, %1, off"
                 :: "v"(lds_off), "v"(ga) : "memory");
#endif
}

// ---------------------------------------------------------------------------
// Pre-pass: bf16 weight buffers, K-chunked (32-wide) layout:
//   WcatC[(k/32)*8192 + n*32 + (k%32)] = Wcat[k][n], Wcat = [W1 ; W2] (512x256)
//   WpC  [(k/32)*8192 + n*32 + (k%32)] = W_pred[n/16][k][n%16]        (256x256)
// A B-fragment lane then reads 32 contiguous bytes from LDS.
// ---------------------------------------------------------------------------
__global__ void prep_weights(const float* __restrict__ W1,
                             const float* __restrict__ W2,
                             const float* __restrict__ Wpred,
                             unsigned short* __restrict__ WcatC,
                             unsigned short* __restrict__ WpC) {
    int idx = blockIdx.x * 256 + threadIdx.x;
    if (idx < 256 * 512) {
        int chunk = idx >> 13, rem = idx & 8191;
        int n = rem >> 5, k = (chunk << 5) | (rem & 31);
        float v = (k < 256) ? W1[k * 256 + n] : W2[(k - 256) * 256 + n];
        WcatC[idx] = f2bf(v);
    }
    int idx2 = idx - 256 * 512;
    if (idx2 >= 0 && idx2 < 256 * 256) {
        int chunk = idx2 >> 13, rem = idx2 & 8191;
        int n = rem >> 5, k = (chunk << 5) | (rem & 31);
        int p = n >> 4, c = n & 15;
        WpC[idx2] = f2bf(Wpred[(p * 256 + k) * 16 + c]);
    }
}

// ---------------------------------------------------------------------------
// Fused GEMM1 + GEMM2 + per-edge loss. 128 threads (4 waves), 64 edges/block.
// Wave w computes rows [w*16, w*16+16) x all 256 output columns.
// ---------------------------------------------------------------------------
__global__ __launch_bounds__(128) void fused_edge_loss(
    const float* __restrict__ h_src, const float* __restrict__ h_dst,
    const float* __restrict__ b_dec, const float* __restrict__ b_pred,
    const int* __restrict__ sidx, const int* __restrict__ didx,
    const int* __restrict__ etc,  const int* __restrict__ emap,
    const unsigned short* __restrict__ WcatC,
    const unsigned short* __restrict__ WpC,
    float* __restrict__ blockSums) {

    __shared__ uint4 s_chunk[1024];              // 16 KB: 32x256 bf16 weight chunk
    __shared__ unsigned short s_h[64 * 256];     // 32 KB: hidden (bf16)
    __shared__ float s_sel[64 * 16];             // 4 KB : selected logits
    __shared__ float s_ps[16], s_pc[16];

    const int tid  = threadIdx.x;
    const int wave = tid >> 5;
    const int lane = tid & 31;
    const int half = lane >> 4;                  // 0: lanes 0-15, 1: lanes 16-31
    const int l    = lane & 15;
    const long blockBase = (long)blockIdx.x * 64;
    const int  rowBase   = wave * 16;

    const unsigned ldsChunkBase = (unsigned)(uintptr_t)&s_chunk[0];

    v8f acc[16];
    v8f vzero = {};
#pragma unroll
    for (int t = 0; t < 16; ++t) acc[t] = vzero;

    // software-pipelined B-fragment fetch (2 deep) to avoid dscnt==0 stalls
    Frag Bf[2];
    auto loadB = [&](int slot, int t) {
        const uint4* bq = s_chunk + ((t * 16 + l) * 4 + half * 2);
        Bf[slot].q[0] = bq[0];
        Bf[slot].q[1] = bq[1];
    };

    // ---------------- GEMM1: K = 512 over [h_src | h_dst] ----------------
    for (int ks = 0; ks < 16; ++ks) {
        const int kk = ks * 32;
        __syncthreads();                          // prior chunk consumers done
#if USE_ASYNC_LDS
        const uint4* wsrc = (const uint4*)WcatC + ks * 1024;
#pragma unroll
        for (int i = 0; i < 8; ++i)
            async_copy16(ldsChunkBase + (unsigned)(i * 128 + tid) * 16u,
                         wsrc + i * 128 + tid);
        wait_async();
#else
        const uint4* wsrc = (const uint4*)WcatC + ks * 1024;
#pragma unroll
        for (int i = 0; i < 8; ++i) s_chunk[i * 128 + tid] = wsrc[i * 128 + tid];
#endif
        __syncthreads();

        // A fragment: 16x32 bf16, contiguous K (lane half h holds k = 16h..16h+15)
        const float* hp = (kk < 256)
            ? (h_src + (blockBase + rowBase + l) * 256 + kk)
            : (h_dst + (blockBase + rowBase + l) * 256 + (kk - 256));
        float4 f0 = *(const float4*)(hp + half * 16);
        float4 f1 = *(const float4*)(hp + half * 16 + 4);
        float4 f2 = *(const float4*)(hp + half * 16 + 8);
        float4 f3 = *(const float4*)(hp + half * 16 + 12);
        Frag A;
        A.u[0] = pk2bf(f0.x, f0.y); A.u[1] = pk2bf(f0.z, f0.w);
        A.u[2] = pk2bf(f1.x, f1.y); A.u[3] = pk2bf(f1.z, f1.w);
        A.u[4] = pk2bf(f2.x, f2.y); A.u[5] = pk2bf(f2.z, f2.w);
        A.u[6] = pk2bf(f3.x, f3.y); A.u[7] = pk2bf(f3.z, f3.w);

        loadB(0, 0);
#pragma unroll
        for (int t = 0; t < 16; ++t) {
            if (t + 1 < 16) loadB((t + 1) & 1, t + 1);
            acc[t] = __builtin_amdgcn_wmma_f32_16x16x32_bf16(
                false, A.v, false, Bf[t & 1].v, (short)0, acc[t], false, false);
        }
    }

    // bias + ReLU, spill h (bf16, row-major) to LDS, reset accumulators
#pragma unroll
    for (int t = 0; t < 16; ++t) {
        float bias = b_dec[t * 16 + l];
#pragma unroll
        for (int v = 0; v < 8; ++v) {
            float x = acc[t][v] + bias;
            x = x > 0.0f ? x : 0.0f;
            int row = rowBase + half * 8 + v;
            s_h[row * 256 + t * 16 + l] = f2bf(x);
            acc[t][v] = 0.0f;
        }
    }

    // ---------------- GEMM2: logits_all = h @ Wp (K = 256) ----------------
    for (int ks = 0; ks < 8; ++ks) {
        const int kk = ks * 32;
        __syncthreads();
#if USE_ASYNC_LDS
        const uint4* wsrc = (const uint4*)WpC + ks * 1024;
#pragma unroll
        for (int i = 0; i < 8; ++i)
            async_copy16(ldsChunkBase + (unsigned)(i * 128 + tid) * 16u,
                         wsrc + i * 128 + tid);
        wait_async();
#else
        const uint4* wsrc = (const uint4*)WpC + ks * 1024;
#pragma unroll
        for (int i = 0; i < 8; ++i) s_chunk[i * 128 + tid] = wsrc[i * 128 + tid];
#endif
        __syncthreads();

        Frag A;
        const unsigned short* hp = s_h + (rowBase + l) * 256 + kk;
        A.q[0] = *(const uint4*)(hp + half * 16);
        A.q[1] = *(const uint4*)(hp + half * 16 + 8);

        loadB(0, 0);
#pragma unroll
        for (int t = 0; t < 16; ++t) {
            if (t + 1 < 16) loadB((t + 1) & 1, t + 1);
            acc[t] = __builtin_amdgcn_wmma_f32_16x16x32_bf16(
                false, A.v, false, Bf[t & 1].v, (short)0, acc[t], false, false);
        }
    }

    // ------- select this edge's pair head (predicated LDS stores) -------
    int pr[8];
#pragma unroll
    for (int v = 0; v < 8; ++v) {
        long e = blockBase + rowBase + half * 8 + v;
        pr[v] = sidx[e] * 4 + didx[e];
    }
#pragma unroll
    for (int t = 0; t < 16; ++t) {
        float bp = b_pred[t * 16 + l];
#pragma unroll
        for (int v = 0; v < 8; ++v) {
            if (pr[v] == t) {
                int row = rowBase + half * 8 + v;
                s_sel[row * 16 + l] = acc[t][v] + bp;
            }
        }
    }
    if (tid < 16) { s_ps[tid] = 0.0f; s_pc[tid] = 0.0f; }
    __syncthreads();

    // ------- per-edge CE loss + per-pair block partial sums -------
    if (tid < 64) {
        long e = blockBase + tid;
        int p   = sidx[e] * 4 + didx[e];
        int lbl = emap[p * 24 + etc[e]];
        float mx = -1e30f;
#pragma unroll
        for (int c = 0; c < 16; ++c) mx = fmaxf(mx, s_sel[tid * 16 + c]);
        float s = 0.0f;
#pragma unroll
        for (int c = 0; c < 16; ++c) s += __expf(s_sel[tid * 16 + c] - mx);
        float loss = (mx + __logf(s)) - s_sel[tid * 16 + lbl];
        atomicAdd(&s_ps[p], loss);
        atomicAdd(&s_pc[p], 1.0f);
    }
    __syncthreads();
    if (tid < 16) {
        blockSums[(long)blockIdx.x * 32 + tid]      = s_ps[tid];
        blockSums[(long)blockIdx.x * 32 + 16 + tid] = s_pc[tid];
    }
}

// ---------------------------------------------------------------------------
// Reduce block partials -> group means -> mean over non-empty groups.
// ---------------------------------------------------------------------------
__global__ __launch_bounds__(256) void finalize_loss(
    const float* __restrict__ blockSums, int nBlocks, float* __restrict__ out) {
    __shared__ float fs[256], fc[256];
    int t = threadIdx.x;
    int p = t & 15, sl = t >> 4;
    float s = 0.0f, c = 0.0f;
    for (int b = sl; b < nBlocks; b += 16) {
        s += blockSums[(long)b * 32 + p];
        c += blockSums[(long)b * 32 + 16 + p];
    }
    fs[t] = s; fc[t] = c;
    __syncthreads();
    if (t < 16) {
        float S = 0.0f, C = 0.0f;
        for (int i = 0; i < 16; ++i) { S += fs[i * 16 + t]; C += fc[i * 16 + t]; }
        fs[t] = (C > 0.0f) ? (S / C) : 0.0f;
        fc[t] = (C > 0.0f) ? 1.0f : 0.0f;
    }
    __syncthreads();
    if (t == 0) {
        float L = 0.0f, P = 0.0f;
        for (int i = 0; i < 16; ++i) { L += fs[i]; P += fc[i]; }
        out[0] = L / P;
    }
}

// ---------------------------------------------------------------------------
extern "C" void kernel_launch(void* const* d_in, const int* in_sizes, int n_in,
                              void* d_out, int out_size, void* d_ws, size_t ws_size,
                              hipStream_t stream) {
    const float* h_src  = (const float*)d_in[0];
    const float* h_dst  = (const float*)d_in[1];
    const float* W1     = (const float*)d_in[2];
    const float* W2     = (const float*)d_in[3];
    const float* b_dec  = (const float*)d_in[4];
    const float* W_pred = (const float*)d_in[5];
    const float* b_pred = (const float*)d_in[6];
    const int*   sidx   = (const int*)d_in[7];
    const int*   didx   = (const int*)d_in[8];
    const int*   etc    = (const int*)d_in[9];
    const int*   emap   = (const int*)d_in[10];
    // d_in[11] = inference flag (reference computes training loss)

    const int E = in_sizes[0] / 256;
    const int nBlocks = E / 64;

    unsigned short* WcatC = (unsigned short*)d_ws;        // 256*512 bf16 = 256 KB
    unsigned short* WpC   = WcatC + 256 * 512;            // 256*256 bf16 = 128 KB
    float* blockSums      = (float*)(WpC + 256 * 256);    // nBlocks*32 f32

    prep_weights<<<768, 256, 0, stream>>>(W1, W2, W_pred, WcatC, WpC);
    fused_edge_loss<<<nBlocks, 128, 0, stream>>>(
        h_src, h_dst, b_dec, b_pred, sidx, didx, etc, emap,
        WcatC, WpC, blockSums);
    finalize_loss<<<1, 256, 0, stream>>>(blockSums, nBlocks, (float*)d_out);
}
